// DevignLite_62577673503033
// MI455X (gfx1250) — matrix operand order repaired and checked by reading, not compiled, not run on hardware
//
#include <hip/hip_runtime.h>
#include <hip/hip_bf16.h>

typedef __attribute__((ext_vector_type(16))) _Float16 v16h;
typedef __attribute__((ext_vector_type(8)))  float    v8f;

#define D 64

// ---------------------------------------------------------------------------
// Degree / normalization
// ---------------------------------------------------------------------------
__global__ void k_init_deg(float* deg, int n) {
    int i = blockIdx.x * blockDim.x + threadIdx.x;
    if (i < n) deg[i] = 1.0f;                    // self-loop contributes 1
}

__global__ void k_count_deg(const int* __restrict__ dst, float* deg, int e) {
    int i = blockIdx.x * blockDim.x + threadIdx.x;
    if (i < e) atomicAdd(&deg[dst[i]], 1.0f);
}

__global__ void k_make_dinv(float* deg, int n) {
    int i = blockIdx.x * blockDim.x + threadIdx.x;
    if (i < n) {
        float d = deg[i];
        deg[i] = (d > 0.0f) ? rsqrtf(d) : 0.0f;  // in place: deg -> dinv
    }
}

// ---------------------------------------------------------------------------
// Embedding gather: x[i][:] = emb[token[i]][:]   (16 lanes/node, float4)
// ---------------------------------------------------------------------------
__global__ void k_gather_emb(const int* __restrict__ tok,
                             const float* __restrict__ emb,
                             float* __restrict__ x, int n) {
    long long tid = (long long)blockIdx.x * blockDim.x + threadIdx.x;
    int i = (int)(tid >> 4);
    if (i >= n) return;
    int q = (int)(tid & 15);
    int t = tok[i];
    ((float4*)(x + (size_t)i * D))[q] =
        ((const float4*)(emb + (size_t)t * D))[q];
}

// ---------------------------------------------------------------------------
// Zero a float4 region
// ---------------------------------------------------------------------------
__global__ void k_zero4(float4* p, long long n4) {
    long long i = (long long)blockIdx.x * blockDim.x + threadIdx.x;
    if (i < n4) p[i] = float4{0.f, 0.f, 0.f, 0.f};
}

// ---------------------------------------------------------------------------
// Helper: pack 16 f32 (four float4) into a v16h fragment
// ---------------------------------------------------------------------------
__device__ __forceinline__ v16h pack16(float4 a, float4 b, float4 c, float4 d) {
    v16h r;
    r[0]  = (_Float16)a.x; r[1]  = (_Float16)a.y;
    r[2]  = (_Float16)a.z; r[3]  = (_Float16)a.w;
    r[4]  = (_Float16)b.x; r[5]  = (_Float16)b.y;
    r[6]  = (_Float16)b.z; r[7]  = (_Float16)b.w;
    r[8]  = (_Float16)c.x; r[9]  = (_Float16)c.y;
    r[10] = (_Float16)c.z; r[11] = (_Float16)c.w;
    r[12] = (_Float16)d.x; r[13] = (_Float16)d.y;
    r[14] = (_Float16)d.z; r[15] = (_Float16)d.w;
    return r;
}

// ---------------------------------------------------------------------------
// WMMA GEMM: Y(N x 64) = X(N x 64) @ W(64 x 64), f16 inputs, f32 accumulate.
// One wave computes one 16-row tile; 8 waves (256 threads) per block.
//
// W is staged into LDS *pre-swizzled into WMMA B-fragment layout*, so each
// lane's 16-element fragment is 32 contiguous bytes (2x ds_load_b128):
//   fragment f = kc*4 + nt   (kc in {0,1}: K-chunk, nt in {0..3}: N-tile)
//   lane l, element i  <->  W[kc*32 + (i>>3)*16 + (l>>4)*8 + (i&7)][nt*16 + (l&15)]
// A fragments (16x32 f16): lane l holds row (l&15); lanes>=16 shift K base
// by 8; elems 0..7 = K base..base+7, elems 8..15 = K base+16..base+23.
// ---------------------------------------------------------------------------
__global__ void k_gemm_wmma(const float* __restrict__ X,
                            const float* __restrict__ W,
                            float* __restrict__ Y, int nrows) {
    __shared__ _Float16 wfrag[8 * 32 * 16];       // 4096 halves = 8 KB
    int t = threadIdx.x;
    #pragma unroll
    for (int i = 0; i < 16; ++i) {
        int idx = t + i * 256;                    // W element (row-major k*64+n)
        int k = idx >> 6, n = idx & 63;
        int kc = k >> 5, lhi = (k >> 3) & 1, ii = (k & 7) | ((k >> 1) & 8);
        int nt = n >> 4, lane = lhi * 16 + (n & 15);
        wfrag[((((kc << 2) | nt) << 5) + lane) * 16 + ii] = (_Float16)W[idx];
    }
    __syncthreads();

    int wave = t >> 5, lane = t & 31;
    int tile = blockIdx.x * 8 + wave;
    if (tile * 16 >= nrows) return;               // wave-uniform exit

    // ---- A fragments: 8x global_load_b128 + pack to f16 ----
    int row = tile * 16 + (lane & 15);
    int q   = (lane >> 4) * 2;                    // float4 index of K base (0|8)/4
    const float4* xr4 = (const float4*)(X + (size_t)row * D);
    float4 p0 = xr4[q + 0], p1 = xr4[q + 1];      // K base .. base+7
    float4 p2 = xr4[q + 4], p3 = xr4[q + 5];      // K base+16 .. base+23
    float4 p4 = xr4[q + 8], p5 = xr4[q + 9];      // +32 chunk
    float4 p6 = xr4[q + 12], p7 = xr4[q + 13];
    v16h a0 = pack16(p0, p1, p2, p3);
    v16h a1 = pack16(p4, p5, p6, p7);

    const v16h* wf = (const v16h*)wfrag;
    int mb = (lane >> 4) * 8;                     // C/D: lanes 16-31 hold M+8

    #pragma unroll
    for (int nt = 0; nt < 4; ++nt) {              // four 16-col output tiles
        int col = nt * 16 + (lane & 15);
        v16h b0 = wf[(0 * 4 + nt) * 32 + lane];   // 2x ds_load_b128 each
        v16h b1 = wf[(1 * 4 + nt) * 32 + lane];
        v8f c = {};
        c = __builtin_amdgcn_wmma_f32_16x16x32_f16(false, a0, false, b0,
                                                   (short)0, c, false, false);
        c = __builtin_amdgcn_wmma_f32_16x16x32_f16(false, a1, false, b1,
                                                   (short)0, c, false, false);
        #pragma unroll
        for (int j = 0; j < 8; ++j)
            Y[(size_t)(tile * 16 + mb + j) * D + col] = c[j];
    }
}

// ---------------------------------------------------------------------------
// Edge scatter: acc[dst] += y[src] * dinv[src]*dinv[dst]  (16 lanes/edge)
// ---------------------------------------------------------------------------
__global__ void k_scatter(const int* __restrict__ src,
                          const int* __restrict__ dst,
                          const float* __restrict__ dinv,
                          const float* __restrict__ Y,
                          float* __restrict__ acc, int e) {
    long long tid = (long long)blockIdx.x * blockDim.x + threadIdx.x;
    int ed = (int)(tid >> 4);
    if (ed >= e) return;
    int q = (int)(tid & 15);
    int s = src[ed], d = dst[ed];
    float nrm = dinv[s] * dinv[d];
    float4 v = ((const float4*)(Y + (size_t)s * D))[q];
    float* out = acc + (size_t)d * D + q * 4;
    atomicAdd(out + 0, v.x * nrm);
    atomicAdd(out + 1, v.y * nrm);
    atomicAdd(out + 2, v.z * nrm);
    atomicAdd(out + 3, v.w * nrm);
}

// ---------------------------------------------------------------------------
// Epilogue: acc = relu(acc + y*dinv^2 + b)   (self-loop + bias + relu)
// ---------------------------------------------------------------------------
__global__ void k_finalize(float* __restrict__ acc,
                           const float* __restrict__ Y,
                           const float* __restrict__ dinv,
                           const float* __restrict__ b, int n) {
    long long tid = (long long)blockIdx.x * blockDim.x + threadIdx.x;
    int i = (int)(tid >> 4);
    if (i >= n) return;
    int q = (int)(tid & 15);
    float d2 = dinv[i] * dinv[i];
    float4 a = ((const float4*)(acc + (size_t)i * D))[q];
    float4 y = ((const float4*)(Y + (size_t)i * D))[q];
    const float* bb = b + q * 4;
    a.x = fmaxf(a.x + y.x * d2 + bb[0], 0.f);
    a.y = fmaxf(a.y + y.y * d2 + bb[1], 0.f);
    a.z = fmaxf(a.z + y.z * d2 + bb[2], 0.f);
    a.w = fmaxf(a.w + y.w * d2 + bb[3], 0.f);
    ((float4*)(acc + (size_t)i * D))[q] = a;
}

// ---------------------------------------------------------------------------
// Pooling: per-graph sum + max (uint-punned atomicMax; values >= 0 post-relu)
// ---------------------------------------------------------------------------
__global__ void k_pool(const float* __restrict__ x,
                       const int* __restrict__ batch,
                       float* __restrict__ h_sum,
                       unsigned int* __restrict__ h_max,
                       float* __restrict__ cnt, int n) {
    long long tid = (long long)blockIdx.x * blockDim.x + threadIdx.x;
    int i = (int)(tid >> 4);
    if (i >= n) return;
    int q = (int)(tid & 15);
    int g = batch[i];
    float4 v = ((const float4*)(x + (size_t)i * D))[q];
    float* s = h_sum + (size_t)g * D + q * 4;
    atomicAdd(s + 0, v.x); atomicAdd(s + 1, v.y);
    atomicAdd(s + 2, v.z); atomicAdd(s + 3, v.w);
    unsigned int* m = h_max + (size_t)g * D + q * 4;
    atomicMax(m + 0, __float_as_uint(v.x));
    atomicMax(m + 1, __float_as_uint(v.y));
    atomicMax(m + 2, __float_as_uint(v.z));
    atomicMax(m + 3, __float_as_uint(v.w));
    if (q == 0) atomicAdd(&cnt[g], 1.0f);
}

// ---------------------------------------------------------------------------
// Classifier head: logits = relu([mean|max] @ Wc1 + bc1) @ Wc2 + bc2
// One block per graph, 64 threads. (4 MFLOP total -> scalar is fine.)
// ---------------------------------------------------------------------------
__global__ void k_classify(const float* __restrict__ h_sum,
                           const float* __restrict__ h_max,
                           const float* __restrict__ cnt,
                           const float* __restrict__ Wc1,
                           const float* __restrict__ bc1,
                           const float* __restrict__ Wc2,
                           const float* __restrict__ bc2,
                           float* __restrict__ out) {
    __shared__ float h[2 * D];
    __shared__ float hid[D];
    int g = blockIdx.x, t = threadIdx.x;
    float c = fmaxf(cnt[g], 1.0f);
    h[t]     = h_sum[(size_t)g * D + t] / c;
    h[D + t] = h_max[(size_t)g * D + t];   // same bits as float
    __syncthreads();
    float a = bc1[t];
    #pragma unroll 4
    for (int k = 0; k < 2 * D; ++k) a += h[k] * Wc1[k * D + t];
    hid[t] = fmaxf(a, 0.f);
    __syncthreads();
    if (t < 2) {
        float o = bc2[t];
        #pragma unroll 4
        for (int d = 0; d < D; ++d) o += hid[d] * Wc2[d * 2 + t];
        out[g * 2 + t] = o;
    }
}

// ---------------------------------------------------------------------------
static inline int divup(long long a, long long b) { return (int)((a + b - 1) / b); }

extern "C" void kernel_launch(void* const* d_in, const int* in_sizes, int n_in,
                              void* d_out, int out_size, void* d_ws, size_t ws_size,
                              hipStream_t stream) {
    const int*   x_tokens = (const int*)  d_in[0];
    const int*   edge_idx = (const int*)  d_in[1];
    const int*   batch    = (const int*)  d_in[2];
    const float* emb      = (const float*)d_in[3];
    const float* Wl[3] = {(const float*)d_in[4], (const float*)d_in[6], (const float*)d_in[8]};
    const float* bl[3] = {(const float*)d_in[5], (const float*)d_in[7], (const float*)d_in[9]};
    const float* Wc1 = (const float*)d_in[10];
    const float* bc1 = (const float*)d_in[11];
    const float* Wc2 = (const float*)d_in[12];
    const float* bc2 = (const float*)d_in[13];

    const int N = in_sizes[0];
    const int E = in_sizes[1] / 2;
    const int G = out_size / 2;
    const int* src = edge_idx;
    const int* dst = edge_idx + E;

    // Workspace layout (256-byte aligned)
    char* ws = (char*)d_ws;
    size_t off = 0;
    auto alloc = [&](size_t bytes) {
        char* p = ws + off;
        off += (bytes + 255) & ~(size_t)255;
        return p;
    };
    float*        dinv  = (float*)alloc((size_t)N * 4);
    float*        xA    = (float*)alloc((size_t)N * D * 4);
    float*        xB    = (float*)alloc((size_t)N * D * 4);
    float*        ybuf  = (float*)alloc((size_t)N * D * 4);
    float*        h_sum = (float*)alloc((size_t)G * D * 4);
    unsigned int* h_max = (unsigned int*)alloc((size_t)G * D * 4);
    float*        cnt   = (float*)alloc((size_t)G * 4);
    (void)ws_size; (void)n_in;

    const int B = 256;

    // 1. degrees -> dinv
    k_init_deg <<<divup(N, B), B, 0, stream>>>(dinv, N);
    k_count_deg<<<divup(E, B), B, 0, stream>>>(dst, dinv, E);
    k_make_dinv<<<divup(N, B), B, 0, stream>>>(dinv, N);

    // 2. embedding gather
    k_gather_emb<<<divup((long long)N * 16, B), B, 0, stream>>>(x_tokens, emb, xA, N);

    // 3. three GCN layers (ping-pong xA/xB; ybuf = x@W)
    float* cur = xA;
    float* nxt = xB;
    const int gemm_blocks = divup(divup(N, 16), 8);
    for (int l = 0; l < 3; ++l) {
        k_gemm_wmma<<<gemm_blocks, 256, 0, stream>>>(cur, Wl[l], ybuf, N);
        k_zero4   <<<divup((long long)N * D / 4, B), B, 0, stream>>>((float4*)nxt, (long long)N * D / 4);
        k_scatter <<<divup((long long)E * 16, B), B, 0, stream>>>(src, dst, dinv, ybuf, nxt, E);
        k_finalize<<<divup((long long)N * 16, B), B, 0, stream>>>(nxt, ybuf, dinv, bl[l], N);
        float* tmp = cur; cur = nxt; nxt = tmp;
    }
    // final activations in `cur`

    // 4. pooling
    k_zero4<<<divup((long long)G * D / 4, B), B, 0, stream>>>((float4*)h_sum, (long long)G * D / 4);
    k_zero4<<<divup((long long)G * D / 4, B), B, 0, stream>>>((float4*)h_max, (long long)G * D / 4);
    k_zero4<<<divup((long long)G / 4, B), B, 0, stream>>>((float4*)cnt, (long long)G / 4);
    k_pool<<<divup((long long)N * 16, B), B, 0, stream>>>(cur, batch, h_sum, h_max, cnt, N);

    // 5. classifier head
    k_classify<<<G, D, 0, stream>>>(h_sum, (const float*)h_max, cnt,
                                    Wc1, bc1, Wc2, bc2, (float*)d_out);
}
